// Attention_14362370637880
// MI455X (gfx1250) — matrix operand order repaired and checked by reading, not compiled, not run on hardware
//
#include <hip/hip_runtime.h>

// ---------------------------------------------------------------------------
// MHA for MI455X (gfx1250): bf16 WMMA GEMMs + flash attention.
// B=16, N=1024, D=512, H=8. All matrix math via v_wmma_f32_16x16x32_bf16.
// Global->LDS staging uses GLOBAL_LOAD_ASYNC_TO_LDS_B128 (ASYNCcnt path).
// ---------------------------------------------------------------------------

typedef __bf16 bf16x16 __attribute__((ext_vector_type(16)));
typedef float  f32x8   __attribute__((ext_vector_type(8)));
typedef float  f32x4   __attribute__((ext_vector_type(4)));
typedef unsigned short u16x8 __attribute__((ext_vector_type(8)));
typedef int    v4i     __attribute__((vector_size(16)));  // GCC-style, matches builtin proto

union ABFrag { bf16x16 v; f32x4 h[2]; };

static constexpr int Bb = 16, Nn = 1024, Dd = 512, Hh = 8;

#if __has_builtin(__builtin_amdgcn_global_load_async_to_lds_b128)
#define USE_ASYNC_LDS 1
#endif

// Copy 16 bytes global -> LDS. Async (ASYNCcnt) path on gfx1250 toolchains
// that expose the builtin; falls back to ld+ds_store otherwise.
__device__ __forceinline__ void cp16_async(const unsigned short* g,
                                           unsigned short* l) {
#ifdef USE_ASYNC_LDS
  __builtin_amdgcn_global_load_async_to_lds_b128(
      (__attribute__((address_space(1))) v4i*)g,
      (__attribute__((address_space(3))) v4i*)l, 0, 0);
#else
  *(f32x4*)l = *(const f32x4*)g;
#endif
}

__device__ __forceinline__ void cp_wait() {
#ifdef USE_ASYNC_LDS
#if __has_builtin(__builtin_amdgcn_s_wait_asynccnt)
  __builtin_amdgcn_s_wait_asynccnt(0);
#else
  asm volatile("s_wait_asynccnt 0" ::: "memory");
#endif
#endif
}

__device__ __forceinline__ unsigned short f2bf(float f) {
  unsigned int u = __float_as_uint(f);
  unsigned int r = (u + 0x7fffu + ((u >> 16) & 1u)) >> 16;  // RNE
  return (unsigned short)r;
}

__device__ __forceinline__ f32x8 zero8() {
  f32x8 z = {0.f, 0.f, 0.f, 0.f, 0.f, 0.f, 0.f, 0.f};
  return z;
}

// ---------------------------------------------------------------------------
// fp32 -> bf16 conversion
// ---------------------------------------------------------------------------
__global__ __launch_bounds__(256) void cvt_f32_bf16(
    const float* __restrict__ x, unsigned short* __restrict__ y, int n) {
  int i = blockIdx.x * 1024 + threadIdx.x;
#pragma unroll
  for (int k = 0; k < 4; ++k) {
    int j = i + k * 256;
    if (j < n) y[j] = f2bf(x[j]);
  }
}

// ---------------------------------------------------------------------------
// Y[M,E] = X[M,K] @ W[E,K]^T + bias[E]   (bf16 inputs, f32 accum)
// Block: 256 thr (8 waves). Block tile 128(M) x 128(E); wave tile 32x64.
// ---------------------------------------------------------------------------
template <bool OUT_F32>
__global__ __launch_bounds__(256) void gemm_xwT(
    const unsigned short* __restrict__ X, const unsigned short* __restrict__ W,
    const float* __restrict__ bias, void* __restrict__ Y, int M, int K, int E) {
  __shared__ alignas(16) unsigned short Xs[128][40];  // +8 pad: bank-safe
  __shared__ alignas(16) unsigned short Ws[128][40];

  const int tid = threadIdx.x;
  const int w = tid >> 5, lane = tid & 31, half = lane >> 4, lr = lane & 15;
  const int wm = w >> 1, we = w & 1;
  const int ebase = blockIdx.x * 128;
  const int mbase = blockIdx.y * 128;

  const int xrow = tid >> 1, xseg = tid & 1;  // 128 rows x 32B each tile

  f32x8 acc[2][4];
#pragma unroll
  for (int i = 0; i < 2; ++i)
#pragma unroll
    for (int j = 0; j < 4; ++j) acc[i][j] = zero8();

  for (int kk = 0; kk < K; kk += 32) {
    const unsigned short* xs = X + (size_t)(mbase + xrow) * K + kk + xseg * 16;
    cp16_async(xs, &Xs[xrow][xseg * 16]);
    cp16_async(xs + 8, &Xs[xrow][xseg * 16 + 8]);
    const unsigned short* wsv = W + (size_t)(ebase + xrow) * K + kk + xseg * 16;
    cp16_async(wsv, &Ws[xrow][xseg * 16]);
    cp16_async(wsv + 8, &Ws[xrow][xseg * 16 + 8]);
    cp_wait();
    __syncthreads();

    ABFrag a[2], bf[4];
#pragma unroll
    for (int i = 0; i < 2; ++i) {  // A 16x32: K chunks at 8*half and 16+8*half
      const int r = 32 * wm + 16 * i + lr;
      a[i].h[0] = *(const f32x4*)&Xs[r][8 * half];
      a[i].h[1] = *(const f32x4*)&Xs[r][16 + 8 * half];
    }
#pragma unroll
    for (int j = 0; j < 4; ++j) {  // B 32x16: K chunks at 8*half and 16+8*half
      const int c = 64 * we + 16 * j + lr;
      bf[j].h[0] = *(const f32x4*)&Ws[c][8 * half];
      bf[j].h[1] = *(const f32x4*)&Ws[c][16 + 8 * half];
    }
#pragma unroll
    for (int i = 0; i < 2; ++i)
#pragma unroll
      for (int j = 0; j < 4; ++j)
        acc[i][j] = __builtin_amdgcn_wmma_f32_16x16x32_bf16(
            false, a[i].v, false, bf[j].v, (short)0, acc[i][j], false, false);
    __syncthreads();
  }

#pragma unroll
  for (int i = 0; i < 2; ++i)
#pragma unroll
    for (int j = 0; j < 4; ++j) {
      const int col = ebase + 64 * we + 16 * j + lr;
      const float bv = bias[col];
#pragma unroll
      for (int r = 0; r < 8; ++r) {  // C layout: row m = r + 8*half
        const size_t row = (size_t)(mbase + 32 * wm + 16 * i + r + 8 * half);
        const float val = acc[i][j][r] + bv;
        if (OUT_F32)
          ((float*)Y)[row * E + col] = val;
        else
          ((unsigned short*)Y)[row * E + col] = f2bf(val);
      }
    }
}

// ---------------------------------------------------------------------------
// Flash attention for one head. Grid (B, N/Br). Block 256 thr (8 waves).
// Br=32 query rows, Bc=32 key rows per step, D=512.
// S phase: 2x2 16x16 tiles x 2 K-halves (8 waves). O: 2x4 wave grid, 16x128 ea.
// Out layout: Out[(b*N+i)*H*D + d*H + h]  (reference's stack(dim=-1) order).
// ---------------------------------------------------------------------------
__global__ __launch_bounds__(256) void attn_head(
    const unsigned short* __restrict__ Qh, const unsigned short* __restrict__ Kh,
    const unsigned short* __restrict__ Vh, unsigned short* __restrict__ Out,
    int h) {
  constexpr int Br = 32, Bc = 32;
  __shared__ alignas(16) unsigned short Qs[2][32][40];
  __shared__ alignas(16) unsigned short Ks[2][32][40];
  __shared__ alignas(16) unsigned short Vt[512][40];  // V transposed [d][j]
  __shared__ alignas(16) float Sb[2][32][36];         // partial scores (K-split)
  __shared__ alignas(16) unsigned short Pb[32][40];   // P tile bf16
  __shared__ float m_s[32], l_s[32], a_s[32];

  const int tid = threadIdx.x;
  const int w = tid >> 5, lane = tid & 31, half = lane >> 4, lr = lane & 15;
  const int ks = w >> 2, srg = (w >> 1) & 1, scg = w & 1;  // S phase
  const int org = w >> 2, ocg = w & 3;                      // O phase
  const int b = blockIdx.x;
  const int ibase = blockIdx.y * Br;
  const float scale = 0.04419417382415922f;  // 1/sqrt(512)

  if (tid < 32) { m_s[tid] = -3.0e38f; l_s[tid] = 0.0f; }
  f32x8 oAcc[8];
#pragma unroll
  for (int t = 0; t < 8; ++t) oAcc[t] = zero8();
  __syncthreads();

  const int sbuf = tid >> 7, srow = (tid >> 2) & 31, sseg = tid & 3;
  const int vj = tid & 31, vdg = tid >> 5;

  for (int jt = 0; jt < Nn / Bc; ++jt) {
    const int jbase = jt * Bc;

    // ---- stage V^T: Vt[d][j] (coalesced b128 reads, unrolled ds scatter) ----
    const unsigned short* vsrc = Vh + (size_t)(b * Nn + jbase + vj) * Dd + vdg * 64;
#pragma unroll
    for (int c8 = 0; c8 < 8; ++c8) {
      u16x8 ch = *(const u16x8*)(vsrc + c8 * 8);
#pragma unroll
      for (int e = 0; e < 8; ++e) Vt[vdg * 64 + c8 * 8 + e][vj] = ch[e];
    }
    // prefetch next K/V tiles into cache (global_prefetch_b8)
    if (jt + 1 < Nn / Bc) {
      const size_t nb = (size_t)(b * Nn + jbase + Bc) * Dd;
      __builtin_prefetch(Kh + nb + (size_t)tid * 64, 0, 1);
      __builtin_prefetch(Vh + nb + (size_t)tid * 64, 0, 1);
    }

    // ---- S = Q K^T, K-dim split across wave groups (256 d each) ----
    f32x8 sAcc = zero8();
    for (int s = 0; s < 8; ++s) {
      const int dq = sbuf * 256 + s * 32;
      cp16_async(Qh + (size_t)(b * Nn + ibase + srow) * Dd + dq + sseg * 8,
                 &Qs[sbuf][srow][sseg * 8]);
      cp16_async(Kh + (size_t)(b * Nn + jbase + srow) * Dd + dq + sseg * 8,
                 &Ks[sbuf][srow][sseg * 8]);
      cp_wait();
      __syncthreads();
      ABFrag aq, bk;
      const int qr = 16 * srg + lr;
      aq.h[0] = *(const f32x4*)&Qs[ks][qr][8 * half];
      aq.h[1] = *(const f32x4*)&Qs[ks][qr][16 + 8 * half];
      const int kc = 16 * scg + lr;  // B col = key row j; K = d chunks at 8*half, 16+8*half
      bk.h[0] = *(const f32x4*)&Ks[ks][kc][8 * half];
      bk.h[1] = *(const f32x4*)&Ks[ks][kc][16 + 8 * half];
      sAcc = __builtin_amdgcn_wmma_f32_16x16x32_bf16(
          false, aq.v, false, bk.v, (short)0, sAcc, false, false);
      __syncthreads();
    }
#pragma unroll
    for (int r = 0; r < 8; ++r)
      Sb[ks][16 * srg + r + 8 * half][16 * scg + lr] = sAcc[r];
    __syncthreads();

    // ---- online softmax (wave 0, one lane per query row) ----
    if (tid < 32) {
      const int row = tid;
      const float mold = m_s[row];
      float mnew = mold;
#pragma unroll
      for (int j = 0; j < Bc; ++j)
        mnew = fmaxf(mnew, (Sb[0][row][j] + Sb[1][row][j]) * scale);
      const float alpha = __expf(mold - mnew);
      float sum = 0.f;
#pragma unroll
      for (int j = 0; j < Bc; ++j) {
        const float p = __expf((Sb[0][row][j] + Sb[1][row][j]) * scale - mnew);
        sum += p;
        Pb[row][j] = f2bf(p);
      }
      m_s[row] = mnew;
      l_s[row] = l_s[row] * alpha + sum;
      a_s[row] = alpha;
    }
    __syncthreads();

    // ---- rescale O, then O += P V ----
    float al[8];
#pragma unroll
    for (int r = 0; r < 8; ++r) al[r] = a_s[16 * org + r + 8 * half];
#pragma unroll
    for (int t = 0; t < 8; ++t)
#pragma unroll
      for (int r = 0; r < 8; ++r) oAcc[t][r] *= al[r];

    ABFrag ap;
    const int pr = 16 * org + lr;
    ap.h[0] = *(const f32x4*)&Pb[pr][8 * half];
    ap.h[1] = *(const f32x4*)&Pb[pr][16 + 8 * half];
#pragma unroll
    for (int t = 0; t < 8; ++t) {
      ABFrag bv;  // B col = d; K = j chunks at 8*half, 16+8*half in Vt rows
      const int dcol = 128 * ocg + 16 * t + lr;
      bv.h[0] = *(const f32x4*)&Vt[dcol][8 * half];
      bv.h[1] = *(const f32x4*)&Vt[dcol][16 + 8 * half];
      oAcc[t] = __builtin_amdgcn_wmma_f32_16x16x32_bf16(
          false, ap.v, false, bv.v, (short)0, oAcc[t], false, false);
    }
    __syncthreads();
  }

  if (tid < 32) a_s[tid] = 1.0f / l_s[tid];
  __syncthreads();

#pragma unroll
  for (int r = 0; r < 8; ++r) {
    const float linv = a_s[16 * org + r + 8 * half];
    const size_t row = (size_t)(b * Nn + ibase + 16 * org + r + 8 * half);
#pragma unroll
    for (int t = 0; t < 8; ++t) {
      const int dcol = 128 * ocg + 16 * t + lr;
      Out[row * (size_t)(Hh * Dd) + (size_t)dcol * Hh + h] = f2bf(oAcc[t][r] * linv);
    }
  }
}

// ---------------------------------------------------------------------------
// Host launcher
// ---------------------------------------------------------------------------
extern "C" void kernel_launch(void* const* d_in, const int* in_sizes, int n_in,
                              void* d_out, int out_size, void* d_ws, size_t ws_size,
                              hipStream_t stream) {
  (void)in_sizes; (void)n_in; (void)out_size; (void)ws_size;
  const float* k_in = (const float*)d_in[0];
  const float* v_in = (const float*)d_in[1];
  const float* q_in = (const float*)d_in[2];
  const float* Wk = (const float*)d_in[3];
  const float* bk = (const float*)d_in[4];
  const float* Wv = (const float*)d_in[5];
  const float* bv = (const float*)d_in[6];
  const float* Wq = (const float*)d_in[7];
  const float* bq = (const float*)d_in[8];
  const float* Wo = (const float*)d_in[9];
  const float* bo = (const float*)d_in[10];

  constexpr size_t SXD = (size_t)Bb * Nn * Dd;   // 8.39M elems
  constexpr size_t SW  = (size_t)Hh * Dd * Dd;   // 2.10M elems
  constexpr size_t SWO = (size_t)Dd * Hh * Dd;   // 2.10M elems
  unsigned short* ws  = (unsigned short*)d_ws;
  unsigned short* qb  = ws;
  unsigned short* kb  = qb + SXD;
  unsigned short* vb  = kb + SXD;
  unsigned short* Wqb = vb + SXD;
  unsigned short* Wkb = Wqb + SW;
  unsigned short* Wvb = Wkb + SW;
  unsigned short* Wob = Wvb + SW;
  unsigned short* qh  = Wob + SWO;   // per-head staging (reused per head)
  unsigned short* kh  = qh + SXD;
  unsigned short* vh  = kh + SXD;
  unsigned short* ao  = vh + SXD;    // [B*N, D*H] interleaved (d,h)

  // fp32 -> bf16 staging
  cvt_f32_bf16<<<(int)(SXD / 1024), 256, 0, stream>>>(q_in, qb, (int)SXD);
  cvt_f32_bf16<<<(int)(SXD / 1024), 256, 0, stream>>>(k_in, kb, (int)SXD);
  cvt_f32_bf16<<<(int)(SXD / 1024), 256, 0, stream>>>(v_in, vb, (int)SXD);
  cvt_f32_bf16<<<(int)(SW / 1024), 256, 0, stream>>>(Wq, Wqb, (int)SW);
  cvt_f32_bf16<<<(int)(SW / 1024), 256, 0, stream>>>(Wk, Wkb, (int)SW);
  cvt_f32_bf16<<<(int)(SW / 1024), 256, 0, stream>>>(Wv, Wvb, (int)SW);
  cvt_f32_bf16<<<(int)(SWO / 1024), 256, 0, stream>>>(Wo, Wob, (int)SWO);

  const int M = Bb * Nn;                  // 16384
  const dim3 ggrid(Dd / 128, M / 128);    // (4, 128)
  for (int h = 0; h < Hh; ++h) {
    gemm_xwT<false><<<ggrid, 256, 0, stream>>>(
        qb, Wqb + (size_t)h * Dd * Dd, bq + (size_t)h * Dd, qh, M, Dd, Dd);
    gemm_xwT<false><<<ggrid, 256, 0, stream>>>(
        kb, Wkb + (size_t)h * Dd * Dd, bk + (size_t)h * Dd, kh, M, Dd, Dd);
    gemm_xwT<false><<<ggrid, 256, 0, stream>>>(
        vb, Wvb + (size_t)h * Dd * Dd, bv + (size_t)h * Dd, vh, M, Dd, Dd);
    attn_head<<<dim3(Bb, Nn / 32), 256, 0, stream>>>(qh, kh, vh, ao, h);
  }
  // final projection: d_out[M, D] = ao[M, H*D] @ Wo^T + bo  (f32 output)
  gemm_xwT<true><<<ggrid, 256, 0, stream>>>(ao, Wob, bo, d_out, M, Hh * Dd, Dd);
}
